// Model_2757369004682
// MI455X (gfx1250) — compile-verified, matrix operand order checked by reading, not run on hardware
//
#include <hip/hip_runtime.h>
#include <hip/hip_bf16.h>

// ---------------------------------------------------------------------------
// VoxelNet-like pipeline for MI455X (gfx1250, wave32, WMMA).
// - VFE2/FCN GEMMs and conv1/conv2 run through v_wmma_f32_16x16x32_f16 (f16 in, f32 acc).
// - Conv weights are staged tap-by-tap into LDS via the Tensor Data Mover
//   (tensor_load_to_lds + s_wait_tensorcnt), read back as ds_load_b128.
// - Next-tap input rows are prefetched (global_prefetch_b8).
// ---------------------------------------------------------------------------

typedef __attribute__((ext_vector_type(16))) _Float16 half16;
typedef __attribute__((ext_vector_type(8)))  _Float16 half8;
typedef __attribute__((ext_vector_type(8)))  float    float8;
typedef __attribute__((ext_vector_type(4)))  unsigned int uint4v;
typedef __attribute__((ext_vector_type(8)))  int      int8v;
typedef __attribute__((ext_vector_type(4)))  int      int4v;

#define N_VOX 20000
#define T_PTS 35
#define FIN   7
#define GX 128
#define GY 128
#define GZ 10
#define NPOS (GX * GY * GZ)          // 163840 (divisible by 16)
#define NT   (N_VOX * T_PTS)         // 700000 (divisible by 16)
#define EPS_BN 1e-5f

#if __has_builtin(__builtin_amdgcn_tensor_load_to_lds) && \
    __has_builtin(__builtin_amdgcn_s_wait_tensorcnt)
#define USE_TDM 1
#else
#define USE_TDM 0
#endif

// ---------------- workspace layout (bytes) ----------------
#define WS_MASK   ((size_t)0)                       // NT floats     (2.8 MB)
#define WS_STATS  ((size_t)(3u << 20))              // 256 floats
#define WS_WPV2   (WS_STATS + 4096)                 // 2048 halfs
#define WS_WPFCN  (WS_WPV2 + 4096)                  // 16384 halfs
#define WS_WPC1   (WS_WPFCN + 32768)                // 221184 halfs
#define WS_WPC2   (WS_WPC1 + 442368)                // 110592 halfs
#define WS_A      ((size_t)(4u << 20))              // 180 MB: h1 / h2 / h3
#define WS_B      (WS_A + (size_t)184 * (1u << 20)) // 180 MB: pwcf1/pwcf2/dense
#define WS_OUT1   (WS_B + (size_t)48 * (1u << 20))
#define WS_OUT2   (WS_B + (size_t)72 * (1u << 20))
#define WS_OUT3   (WS_B + (size_t)96 * (1u << 20))

// ---------------- small utility kernels ----------------

__global__ void zero_f32_kernel(float* p, long n) {
  long i = (long)blockIdx.x * blockDim.x + threadIdx.x;
  if (i < n) p[i] = 0.f;
}

__global__ void zero_f16_kernel(_Float16* p, long n) {
  long i = (long)blockIdx.x * blockDim.x + threadIdx.x;
  if (i < n) p[i] = (_Float16)0.f;
}

// Per-channel sum / sum-of-squares over a row-major [rows][ch] f16 buffer.
__global__ void stats_f16_kernel(const _Float16* __restrict__ d, long rows,
                                 int ch, float* __restrict__ acc) {
  int c = blockIdx.y;
  __shared__ float s1[256];
  __shared__ float s2[256];
  float a = 0.f, b = 0.f;
  for (long r = (long)blockIdx.x * blockDim.x + threadIdx.x; r < rows;
       r += (long)gridDim.x * blockDim.x) {
    float v = (float)d[r * ch + c];
    a += v;
    b += v * v;
  }
  s1[threadIdx.x] = a;
  s2[threadIdx.x] = b;
  __syncthreads();
  for (int s = 128; s > 0; s >>= 1) {
    if ((int)threadIdx.x < s) {
      s1[threadIdx.x] += s1[threadIdx.x + s];
      s2[threadIdx.x] += s2[threadIdx.x + s];
    }
    __syncthreads();
  }
  if (threadIdx.x == 0) {
    atomicAdd(&acc[c], s1[0]);
    atomicAdd(&acc[ch + c], s2[0]);
  }
}

// acc[c] <- mean, acc[ch+c] <- rsqrt(var + eps)
__global__ void finalize_stats_kernel(float* acc, int ch, float count) {
  int c = threadIdx.x;
  if (c < ch) {
    float m = acc[c] / count;
    float v = acc[ch + c] / count - m * m;
    acc[c] = m;
    acc[ch + c] = rsqrtf(v + EPS_BN);
  }
}

// In-place BN + ReLU over row-major [*][ch] f16 buffer.
__global__ void bnrelu_f16_kernel(_Float16* d, long n, int ch,
                                  const float* __restrict__ acc,
                                  const float* __restrict__ g,
                                  const float* __restrict__ be) {
  long i = (long)blockIdx.x * blockDim.x + threadIdx.x;
  if (i >= n) return;
  int c = (int)(i % ch);
  float v = (float)d[i];
  v = (v - acc[c]) * acc[ch + c] * g[c] + be[c];
  d[i] = (_Float16)fmaxf(v, 0.f);
}

// ---------------- stage 1: VFE1 pointwise (7 -> 16) + mask ----------------

__global__ void vfe1_kernel(const float* __restrict__ vf,
                            const float* __restrict__ w,
                            const float* __restrict__ b,
                            _Float16* __restrict__ h1,
                            float* __restrict__ mask) {
  long r = (long)blockIdx.x * blockDim.x + threadIdx.x;
  if (r >= NT) return;
  float x[FIN];
  float mx = -3.4e38f;
#pragma unroll
  for (int i = 0; i < FIN; ++i) {
    x[i] = vf[r * FIN + i];
    mx = fmaxf(mx, x[i]);
  }
  mask[r] = (mx != 0.f) ? 1.f : 0.f;
#pragma unroll
  for (int c = 0; c < 16; ++c) {
    float a = b[c];
#pragma unroll
    for (int i = 0; i < FIN; ++i) a += x[i] * w[i * 16 + c];
    h1[r * 16 + c] = (_Float16)a;
  }
}

// BN+ReLU on h, max over T, concat [pwf | laf], apply mask -> f16 [NT][2C].
__global__ void vfe_post_kernel(const _Float16* __restrict__ h,
                                const float* __restrict__ acc,
                                const float* __restrict__ g,
                                const float* __restrict__ be,
                                const float* __restrict__ mask,
                                _Float16* __restrict__ pwcf, int C) {
  long id = (long)blockIdx.x * blockDim.x + threadIdx.x;
  long n = id / C;
  int c = (int)(id % C);
  if (n >= N_VOX) return;
  float mean = acc[c], inv = acc[C + c], gg = g[c], bb = be[c];
  float mx = 0.f;
  for (int t = 0; t < T_PTS; ++t) {
    long row = n * T_PTS + t;
    float v = (float)h[row * C + c];
    v = fmaxf((v - mean) * inv * gg + bb, 0.f);
    pwcf[row * (2 * C) + c] = (_Float16)(v * mask[row]);
    mx = fmaxf(mx, v);
  }
  for (int t = 0; t < T_PTS; ++t) {
    long row = n * T_PTS + t;
    pwcf[row * (2 * C) + C + c] = (_Float16)(mx * mask[row]);
  }
}

// ---------------- weight packing into WMMA fragment order ----------------
// Fragment (16-bit, K-dim): lane l, koff = (l>=16)?8:0
//   halfs j=0..7  -> K = base + koff + j
//   halfs j=8..15 -> K = base + 16 + koff + (j-8)

__global__ void pack_w_kernel(const float* __restrict__ w,  // [K][N] row-major
                              _Float16* __restrict__ wp, int K, int N) {
  int id = blockIdx.x * blockDim.x + threadIdx.x;
  int ksteps = K >> 5;
  int total = (N >> 4) * ksteps * 512;
  if (id >= total) return;
  int j = id & 15, l = (id >> 4) & 31;
  int rest = id >> 9;
  int ks = rest % ksteps, nt = rest / ksteps;
  int koff = (l >> 4) ? 8 : 0;
  int k = ks * 32 + koff + (j < 8 ? j : j + 8);
  int n = (nt << 4) + (l & 15);
  wp[id] = (_Float16)w[k * N + n];
}

// Conv weights [O][I][27] -> A-fragments per (tap, kstep, mtile).
__global__ void pack_cw_kernel(const float* __restrict__ w,
                               _Float16* __restrict__ wp, int O, int I) {
  int id = blockIdx.x * blockDim.x + threadIdx.x;
  int ksteps = I >> 5, mtiles = O >> 4;
  int total = 27 * ksteps * mtiles * 512;
  if (id >= total) return;
  int j = id & 15, l = (id >> 4) & 31;
  int rest = id >> 9;
  int mt = rest % mtiles; rest /= mtiles;
  int ks = rest % ksteps;
  int tap = rest / ksteps;
  int koff = (l >> 4) ? 8 : 0;
  int k = ks * 32 + koff + (j < 8 ? j : j + 8);
  int m = (mt << 4) + (l & 15);
  wp[id] = (_Float16)w[((long)m * I + k) * 27 + tap];
}

// ---------------- WMMA GEMM: out[rows][N] = A[rows][K] @ W + bias ----------

template <int NTILES, int KSTEPS>
__global__ void gemm_wmma_kernel(const _Float16* __restrict__ A,
                                 const _Float16* __restrict__ Wp,
                                 const float* __restrict__ bias,
                                 _Float16* __restrict__ out, int mtiles) {
  const int K = KSTEPS * 32, N = NTILES * 16;
  int wave = (int)((blockIdx.x * blockDim.x + threadIdx.x) >> 5);
  int lane = threadIdx.x & 31;
  if (wave >= mtiles) return;
  int koff = (lane >> 4) ? 8 : 0;
  long m = ((long)wave << 4) + (lane & 15);

  float8 acc[NTILES];
#pragma unroll
  for (int i = 0; i < NTILES; ++i)
#pragma unroll
    for (int j = 0; j < 8; ++j) acc[i][j] = 0.f;

#pragma unroll
  for (int ks = 0; ks < KSTEPS; ++ks) {
    const _Float16* arow = A + m * K + ks * 32 + koff;
    half8 alo = *(const half8*)(arow);
    half8 ahi = *(const half8*)(arow + 16);
    half16 a = __builtin_shufflevector(alo, ahi, 0, 1, 2, 3, 4, 5, 6, 7, 8, 9,
                                       10, 11, 12, 13, 14, 15);
#pragma unroll
    for (int nt = 0; nt < NTILES; ++nt) {
      const _Float16* wrow = Wp + (((long)nt * KSTEPS + ks) * 32 + lane) * 16;
      half8 blo = *(const half8*)(wrow);
      half8 bhi = *(const half8*)(wrow + 8);
      half16 b = __builtin_shufflevector(blo, bhi, 0, 1, 2, 3, 4, 5, 6, 7, 8,
                                         9, 10, 11, 12, 13, 14, 15);
      acc[nt] = __builtin_amdgcn_wmma_f32_16x16x32_f16(
          false, a, false, b, (short)0, acc[nt], false, false);
    }
  }

  // D layout: VGPR r -> M = r + (lane>=16 ? 8 : 0), N = lane & 15
  long mrow0 = ((long)wave << 4) + ((lane >> 4) << 3);
  int col = lane & 15;
#pragma unroll
  for (int nt = 0; nt < NTILES; ++nt) {
    int n = (nt << 4) + col;
    float bz = bias[n];
#pragma unroll
    for (int r = 0; r < 8; ++r)
      out[(mrow0 + r) * N + n] = (_Float16)(acc[nt][r] + bz);
  }
}

// ---------------- scatter into dense grid ----------------

__global__ void fcn_post_scatter_kernel(const _Float16* __restrict__ h3,
                                        const float* __restrict__ acc,
                                        const float* __restrict__ g,
                                        const float* __restrict__ be,
                                        const int* __restrict__ coords,
                                        _Float16* __restrict__ dense) {
  long id = (long)blockIdx.x * blockDim.x + threadIdx.x;
  long n = id >> 7;
  int c = (int)(id & 127);
  if (n >= N_VOX) return;
  float mean = acc[c], inv = acc[128 + c], gg = g[c], bb = be[c];
  float mx = 0.f;
  for (int t = 0; t < T_PTS; ++t) {
    float v = (float)h3[(n * T_PTS + t) * 128 + c];
    v = fmaxf((v - mean) * inv * gg + bb, 0.f);
    mx = fmaxf(mx, v);
  }
  int cx = coords[n * 4 + 1], cy = coords[n * 4 + 2], cz = coords[n * 4 + 3];
  long idx = ((long)cx * GY + cy) * GZ + cz;
  dense[idx * 128 + c] = (_Float16)mx;
}

// ---------------- TDM helper: stage one tap's packed weights into LDS ------
// D# per ISA ch.8: group0 {count=1, lds_addr, global_addr, type=2},
// group1 {data_size=2B, tensor_dim0 = nelem, dim1 = 1, tile_dim0 = nelem,
//         tile_dim1 = 1, dim0_stride = nelem}. Groups 2/3 zero (<=2D tensor).
__device__ __forceinline__ void tdm_stage_weights(const _Float16* gsrc,
                                                  unsigned lds_off,
                                                  int nelem) {
#if USE_TDM
  unsigned long long ga = (unsigned long long)(const void*)gsrc;
  uint4v g0;
  g0[0] = 1u;                                   // count=1, user descriptor
  g0[1] = lds_off;                              // LDS byte address
  g0[2] = (unsigned)ga;                         // global_addr[31:0]
  g0[3] = (unsigned)((ga >> 32) & 0x1FFFFFFu) | (2u << 30);  // addr hi | type=2
  unsigned ne = (unsigned)nelem;
  int8v g1;
  g1[0] = (int)(1u << 16);                      // data_size = 1 -> 2 bytes
  g1[1] = (int)((ne & 0xFFFFu) << 16);          // tensor_dim0[15:0]  @bits 63:48
  g1[2] = (int)((ne >> 16) | (1u << 16));       // dim0[31:16], tensor_dim1 = 1
  g1[3] = (int)((ne & 0xFFFFu) << 16);          // tile_dim0 @bits 127:112
  g1[4] = 1;                                    // tile_dim1 = 1
  g1[5] = (int)ne;                              // tensor_dim0_stride[31:0]
  g1[6] = 0;
  g1[7] = 0;
  int4v gz;
  gz[0] = 0; gz[1] = 0; gz[2] = 0; gz[3] = 0;
  if (threadIdx.x < 32) {                       // one TDM issue per workgroup
#if __clang_major__ >= 23
    int8v gz8;
    gz8[0] = 0; gz8[1] = 0; gz8[2] = 0; gz8[3] = 0;
    gz8[4] = 0; gz8[5] = 0; gz8[6] = 0; gz8[7] = 0;
    __builtin_amdgcn_tensor_load_to_lds(g0, g1, gz, gz, gz8, 0);
#else
    __builtin_amdgcn_tensor_load_to_lds(g0, g1, gz, gz, 0);
#endif
  }
  __builtin_amdgcn_s_wait_tensorcnt(0);         // no-op for non-issuing waves
#else
  (void)lds_off;
#endif
}

// ---------------- implicit-GEMM 3x3x3 conv via WMMA + TDM staging ----------
// in:  [GX][GY][GZ][IC] f16 (channel-minor), out: [pos][OC] f16 (+bias).
// One wave = 16 spatial positions x OC channels. 8 waves/block share the
// per-tap weight block staged once into LDS. Grid divides exactly, so all
// waves participate in every barrier.

template <int ICSTEPS, int OCTILES>
__global__ __launch_bounds__(256) void conv_wmma_kernel(
    const _Float16* __restrict__ in, const _Float16* __restrict__ Wp,
    const float* __restrict__ bias, _Float16* __restrict__ out) {
  const int IC = ICSTEPS * 32, OC = OCTILES * 16;
  const int WELEM = ICSTEPS * OCTILES * 512;      // halfs per tap
  __shared__ _Float16 wlds[WELEM];                // 16 KB (c1) / 8 KB (c2)

  int wave = (int)((blockIdx.x * blockDim.x + threadIdx.x) >> 5);
  int lane = threadIdx.x & 31;
  int p = (wave << 4) + (lane & 15);
  int x = p / (GY * GZ);
  int rem = p % (GY * GZ);
  int y = rem / GZ;
  int z = rem % GZ;
  int koff = (lane >> 4) ? 8 : 0;

  float8 acc[OCTILES];
#pragma unroll
  for (int i = 0; i < OCTILES; ++i)
#pragma unroll
    for (int j = 0; j < 8; ++j) acc[i][j] = 0.f;

  for (int tap = 0; tap < 27; ++tap) {
    __syncthreads();  // previous tap's LDS reads complete
    const _Float16* wsrc = Wp + (long)tap * WELEM;
#if USE_TDM
    tdm_stage_weights(wsrc, 0u /* wlds is the only (base) LDS allocation */,
                      WELEM);
#else
    for (int i = threadIdx.x; i < WELEM / 8; i += 256)
      ((half8*)wlds)[i] = ((const half8*)wsrc)[i];
#endif
    __syncthreads();  // weights visible to all waves

    int dx = tap / 9 - 1, dy = (tap / 3) % 3 - 1, dz = tap % 3 - 1;
    int xx = x + dx, yy = y + dy, zz = z + dz;
    bool ok = (unsigned)xx < GX && (unsigned)yy < GY && (unsigned)zz < GZ;
    const _Float16* row = in + ((long)(xx * GY + yy) * GZ + zz) * IC;

    // prefetch next tap's input row (L2-resident dense grid)
    if (tap + 1 < 27) {
      int t2 = tap + 1;
      int x2 = x + t2 / 9 - 1, y2 = y + (t2 / 3) % 3 - 1, z2 = z + t2 % 3 - 1;
      if ((unsigned)x2 < GX && (unsigned)y2 < GY && (unsigned)z2 < GZ)
        __builtin_prefetch(in + ((long)(x2 * GY + y2) * GZ + z2) * IC, 0, 1);
    }

#pragma unroll
    for (int ks = 0; ks < ICSTEPS; ++ks) {
      half16 bf;
      if (ok) {
        half8 blo = *(const half8*)(row + ks * 32 + koff);
        half8 bhi = *(const half8*)(row + ks * 32 + koff + 16);
        bf = __builtin_shufflevector(blo, bhi, 0, 1, 2, 3, 4, 5, 6, 7, 8, 9,
                                     10, 11, 12, 13, 14, 15);
      } else {
#pragma unroll
        for (int i = 0; i < 16; ++i) bf[i] = (_Float16)0.f;
      }
#pragma unroll
      for (int mt = 0; mt < OCTILES; ++mt) {
        const _Float16* wrow = wlds + (ks * OCTILES + mt) * 512 + lane * 16;
        half8 alo = *(const half8*)(wrow);          // ds_load_b128
        half8 ahi = *(const half8*)(wrow + 8);      // ds_load_b128
        half16 af = __builtin_shufflevector(alo, ahi, 0, 1, 2, 3, 4, 5, 6, 7,
                                            8, 9, 10, 11, 12, 13, 14, 15);
        acc[mt] = __builtin_amdgcn_wmma_f32_16x16x32_f16(
            false, af, false, bf, (short)0, acc[mt], false, false);
      }
    }
  }

  // D: M = out-channel (r + 8*(lane>=16)), N = position (lane & 15)
  long pbase = ((long)wave << 4) + (lane & 15);
#pragma unroll
  for (int mt = 0; mt < OCTILES; ++mt) {
    int oc0 = (mt << 4) + ((lane >> 4) << 3);
#pragma unroll
    for (int r = 0; r < 8; ++r) {
      int oc = oc0 + r;
      out[pbase * OC + oc] = (_Float16)(acc[mt][r] + bias[oc]);
    }
  }
}

// ---------------- conv3 (64 -> 2 channels): plain VALU ----------------

__global__ void conv3_kernel(const _Float16* __restrict__ in,  // [pos][64]
                             const float* __restrict__ w,      // [2][64][27]
                             const float* __restrict__ bias,
                             _Float16* __restrict__ out) {     // [pos][2]
  int p = blockIdx.x * blockDim.x + threadIdx.x;
  if (p >= NPOS) return;
  int x = p / (GY * GZ);
  int rem = p % (GY * GZ);
  int y = rem / GZ;
  int z = rem % GZ;
  float a0 = bias[0], a1 = bias[1];
  for (int tap = 0; tap < 27; ++tap) {
    int dx = tap / 9 - 1, dy = (tap / 3) % 3 - 1, dz = tap % 3 - 1;
    int xx = x + dx, yy = y + dy, zz = z + dz;
    if ((unsigned)xx >= GX || (unsigned)yy >= GY || (unsigned)zz >= GZ)
      continue;
    const _Float16* row = in + ((long)(xx * GY + yy) * GZ + zz) * 64;
    for (int ic = 0; ic < 64; ++ic) {
      float v = (float)row[ic];
      a0 += v * w[(0 * 64 + ic) * 27 + tap];
      a1 += v * w[(1 * 64 + ic) * 27 + tap];
    }
  }
  out[(long)p * 2 + 0] = (_Float16)a0;
  out[(long)p * 2 + 1] = (_Float16)a1;
}

// ---------------- gather + softmax ----------------

__global__ void final_softmax_kernel(const _Float16* __restrict__ y,
                                     const int* __restrict__ coords,
                                     float* __restrict__ out) {
  int v = blockIdx.x * blockDim.x + threadIdx.x;
  if (v >= N_VOX) return;
  int cx = coords[v * 4 + 1], cy = coords[v * 4 + 2], cz = coords[v * 4 + 3];
  long idx = ((long)cx * GY + cy) * GZ + cz;
  float a = (float)y[idx * 2 + 0];
  float b = (float)y[idx * 2 + 1];
  float m = fmaxf(a, b);
  float e0 = expf(a - m), e1 = expf(b - m);
  float s = e0 + e1;
  out[v * 2 + 0] = e0 / s;
  out[v * 2 + 1] = e1 / s;
}

// ---------------------------------------------------------------------------

extern "C" void kernel_launch(void* const* d_in, const int* in_sizes, int n_in,
                              void* d_out, int out_size, void* d_ws,
                              size_t ws_size, hipStream_t stream) {
  const float* vf      = (const float*)d_in[0];
  const int*   coords  = (const int*)d_in[1];
  const float* vfe1_w  = (const float*)d_in[2];
  const float* vfe1_b  = (const float*)d_in[3];
  const float* vfe1_g  = (const float*)d_in[4];
  const float* vfe1_be = (const float*)d_in[5];
  const float* vfe2_w  = (const float*)d_in[6];
  const float* vfe2_b  = (const float*)d_in[7];
  const float* vfe2_g  = (const float*)d_in[8];
  const float* vfe2_be = (const float*)d_in[9];
  const float* fcn_w   = (const float*)d_in[10];
  const float* fcn_b   = (const float*)d_in[11];
  const float* fcn_g   = (const float*)d_in[12];
  const float* fcn_be  = (const float*)d_in[13];
  const float* c1_w    = (const float*)d_in[14];
  const float* c1_b    = (const float*)d_in[15];
  const float* c1_g    = (const float*)d_in[16];
  const float* c1_be   = (const float*)d_in[17];
  const float* c2_w    = (const float*)d_in[18];
  const float* c2_b    = (const float*)d_in[19];
  const float* c2_g    = (const float*)d_in[20];
  const float* c2_be   = (const float*)d_in[21];
  const float* c3_w    = (const float*)d_in[22];
  const float* c3_b    = (const float*)d_in[23];
  const float* c3_g    = (const float*)d_in[24];
  const float* c3_be   = (const float*)d_in[25];

  char* ws = (char*)d_ws;
  float*     mask  = (float*)(ws + WS_MASK);
  float*     stats = (float*)(ws + WS_STATS);
  _Float16*  wpv2  = (_Float16*)(ws + WS_WPV2);
  _Float16*  wpfcn = (_Float16*)(ws + WS_WPFCN);
  _Float16*  wpc1  = (_Float16*)(ws + WS_WPC1);
  _Float16*  wpc2  = (_Float16*)(ws + WS_WPC2);
  _Float16*  bufA  = (_Float16*)(ws + WS_A);   // h1 / h2 / h3
  _Float16*  bufB  = (_Float16*)(ws + WS_B);   // pwcf1 / pwcf2 / dense
  _Float16*  dense = (_Float16*)(ws + WS_B);
  _Float16*  out1  = (_Float16*)(ws + WS_OUT1);
  _Float16*  out2  = (_Float16*)(ws + WS_OUT2);
  _Float16*  out3  = (_Float16*)(ws + WS_OUT3);
  float*     outp  = (float*)d_out;

  const int B256 = 256;
  const int MTILES = NT / 16;        // 43750
  const int GEMM_BLOCKS = (MTILES * 32 + B256 - 1) / B256;
  const int CONV_BLOCKS = (NPOS / 16) / 8;  // 1280, exact

  // weight packing (fragment layouts)
  pack_w_kernel<<<(2048 + 255) / 256, B256, 0, stream>>>(vfe2_w, wpv2, 32, 64);
  pack_w_kernel<<<(16384 + 255) / 256, B256, 0, stream>>>(fcn_w, wpfcn, 128, 128);
  pack_cw_kernel<<<(221184 + 255) / 256, B256, 0, stream>>>(c1_w, wpc1, 64, 128);
  pack_cw_kernel<<<(110592 + 255) / 256, B256, 0, stream>>>(c2_w, wpc2, 64, 64);

  // ---- VFE1: linear -> stats -> BN/ReLU/max/concat/mask ----
  vfe1_kernel<<<(NT + 255) / 256, B256, 0, stream>>>(vf, vfe1_w, vfe1_b, bufA, mask);
  zero_f32_kernel<<<1, B256, 0, stream>>>(stats, 256);
  stats_f16_kernel<<<dim3(64, 16), B256, 0, stream>>>(bufA, NT, 16, stats);
  finalize_stats_kernel<<<1, B256, 0, stream>>>(stats, 16, (float)NT);
  vfe_post_kernel<<<(N_VOX * 16 + 255) / 256, B256, 0, stream>>>(
      bufA, stats, vfe1_g, vfe1_be, mask, bufB, 16);

  // ---- VFE2: WMMA GEMM [NT,32]@[32,64] -> stats -> post ----
  gemm_wmma_kernel<4, 1><<<GEMM_BLOCKS, B256, 0, stream>>>(bufB, wpv2, vfe2_b,
                                                           bufA, MTILES);
  zero_f32_kernel<<<1, B256, 0, stream>>>(stats, 256);
  stats_f16_kernel<<<dim3(64, 64), B256, 0, stream>>>(bufA, NT, 64, stats);
  finalize_stats_kernel<<<1, B256, 0, stream>>>(stats, 64, (float)NT);
  vfe_post_kernel<<<(N_VOX * 64 + 255) / 256, B256, 0, stream>>>(
      bufA, stats, vfe2_g, vfe2_be, mask, bufB, 64);

  // ---- FCN: WMMA GEMM [NT,128]@[128,128] -> stats -> max/scatter ----
  gemm_wmma_kernel<8, 4><<<GEMM_BLOCKS, B256, 0, stream>>>(bufB, wpfcn, fcn_b,
                                                           bufA, MTILES);
  zero_f32_kernel<<<1, B256, 0, stream>>>(stats, 256);
  stats_f16_kernel<<<dim3(64, 128), B256, 0, stream>>>(bufA, NT, 128, stats);
  finalize_stats_kernel<<<1, B256, 0, stream>>>(stats, 128, (float)NT);
  zero_f16_kernel<<<((long)NPOS * 128 + 255) / 256, B256, 0, stream>>>(
      dense, (long)NPOS * 128);
  fcn_post_scatter_kernel<<<(N_VOX * 128 + 255) / 256, B256, 0, stream>>>(
      bufA, stats, fcn_g, fcn_be, coords, dense);

  // ---- conv1 (128->64) implicit-GEMM WMMA + BN/ReLU ----
  conv_wmma_kernel<4, 4><<<CONV_BLOCKS, B256, 0, stream>>>(dense, wpc1, c1_b,
                                                           out1);
  zero_f32_kernel<<<1, B256, 0, stream>>>(stats, 256);
  stats_f16_kernel<<<dim3(64, 64), B256, 0, stream>>>(out1, NPOS, 64, stats);
  finalize_stats_kernel<<<1, B256, 0, stream>>>(stats, 64, (float)NPOS);
  bnrelu_f16_kernel<<<((long)NPOS * 64 + 255) / 256, B256, 0, stream>>>(
      out1, (long)NPOS * 64, 64, stats, c1_g, c1_be);

  // ---- conv2 (64->64) implicit-GEMM WMMA + BN/ReLU ----
  conv_wmma_kernel<2, 4><<<CONV_BLOCKS, B256, 0, stream>>>(out1, wpc2, c2_b,
                                                           out2);
  zero_f32_kernel<<<1, B256, 0, stream>>>(stats, 256);
  stats_f16_kernel<<<dim3(64, 64), B256, 0, stream>>>(out2, NPOS, 64, stats);
  finalize_stats_kernel<<<1, B256, 0, stream>>>(stats, 64, (float)NPOS);
  bnrelu_f16_kernel<<<((long)NPOS * 64 + 255) / 256, B256, 0, stream>>>(
      out2, (long)NPOS * 64, 64, stats, c2_g, c2_be);

  // ---- conv3 (64->2) + BN/ReLU ----
  conv3_kernel<<<(NPOS + 255) / 256, B256, 0, stream>>>(out2, c3_w, c3_b, out3);
  zero_f32_kernel<<<1, B256, 0, stream>>>(stats, 256);
  stats_f16_kernel<<<dim3(64, 2), B256, 0, stream>>>(out3, NPOS, 2, stats);
  finalize_stats_kernel<<<1, B256, 0, stream>>>(stats, 2, (float)NPOS);
  bnrelu_f16_kernel<<<((long)NPOS * 2 + 255) / 256, B256, 0, stream>>>(
      out3, (long)NPOS * 2, 2, stats, c3_g, c3_be);

  // ---- gather + softmax ----
  final_softmax_kernel<<<(N_VOX + 255) / 256, B256, 0, stream>>>(out3, coords,
                                                                 outp);
}
